// LSTMCellModel_29815662969161
// MI455X (gfx1250) — matrix operand order repaired and checked by reading, not compiled, not run on hardware
//
#include <hip/hip_runtime.h>
#include <hip/hip_bf16.h>

// Problem constants (match reference)
#define B_  256
#define S_  512
#define I_  128
#define H_  256
#define O_  64

typedef __bf16 bf16;
typedef __attribute__((ext_vector_type(16))) __bf16 v16bf;
typedef __attribute__((ext_vector_type(8)))  __bf16 v8bf;
typedef __attribute__((ext_vector_type(8)))  float  v8f;

// Workspace layout (bytes): bf16-packed WMMA B-operands
#define WIH_PK_OFF  0                             // 1024 x 128 bf16 = 256 KB
#define WHH_PK_OFF  (1024 * 128 * 2)              // 1024 x 256 bf16 = 512 KB
#define WOUT_PK_OFF (WHH_PK_OFF + 1024 * 256 * 2) // 64 x 256 bf16 = 32 KB

// Packed-element strides (in ushort elements)
#define GIH_STRIDE 32768   // 16 tiles * 4 ksteps * 32 lanes * 16 : gate stride in wih_pk
#define GHH_STRIDE 65536   // 16 tiles * 8 ksteps * 32 lanes * 16 : gate stride in whh_pk
#define KK_STRIDE  512     // 32 lanes * 16 elems per (tile, kstep) block

#define XS_ROW 136         // x LDS row stride in bf16 (128 + 8 pad)
#define HS_ROW 264         // h LDS row stride in bf16 (256 + 8 pad)

// ---------------------------------------------------------------------------
// Pack a row-major f32 weight matrix W[ntiles*16][K] into bf16 in the exact
// per-lane k order this file uses for the WMMA 16x16x32 B operand (contiguous k; A matches):
//   lane = (s<<4)|n16 holds column n = tile*16+n16, contiguous K chunk
//   k = kk*32 + s*16 + e (e = 0..15).  B[k][n] = W[n][k] (gates = act @ W^T),
//   so each lane reads a contiguous 16-float chunk of row n of W.
// Packed flat index: (((ntile*ksteps + kk)*32 + lane)*16 + e).
// ---------------------------------------------------------------------------
__global__ void pack_b_operand(const float* __restrict__ W,
                               unsigned short* __restrict__ out,
                               int K, int ksteps, int total_threads) {
  const int tid = blockIdx.x * blockDim.x + threadIdx.x;
  if (tid >= total_threads) return;
  const int lane = tid & 31;
  const int kk   = (tid >> 5) % ksteps;
  const int j    = tid / (ksteps * 32);
  const int n    = j * 16 + (lane & 15);
  const int s    = lane >> 4;
  const float* wrow = W + (size_t)n * K + kk * 32 + s * 16;
  unsigned short* o = out + (size_t)tid * 16;
#pragma unroll
  for (int e = 0; e < 16; ++e) {
    union { __bf16 b; unsigned short u; } cv;
    cv.b = (__bf16)wrow[e];
    o[e] = cv.u;
  }
}

// A operand (16x32 bf16) per-lane gather from a row-major LDS row:
// lane (m, s): elems 0..15 -> K = kk*32 + s*16 + e (contiguous, matches B pack).
__device__ __forceinline__ v16bf load_a_tile(const bf16* rowbase, int kk, int s) {
  const v8bf lo = *(const v8bf*)(rowbase + kk * 32 + s * 16);
  const v8bf hi = *(const v8bf*)(rowbase + kk * 32 + 8 + s * 16);
  return __builtin_shufflevector(lo, hi, 0, 1, 2, 3, 4, 5, 6, 7,
                                         8, 9, 10, 11, 12, 13, 14, 15);
}

#define LOG2E 1.4426950408889634f
// One v_exp_f32 + one v_rcp_f32 each.
__device__ __forceinline__ float sigf(float x) {
  return __builtin_amdgcn_rcpf(1.0f + __builtin_amdgcn_exp2f(-LOG2E * x));
}
__device__ __forceinline__ float tanh_fast(float x) {
  return 2.0f * __builtin_amdgcn_rcpf(1.0f + __builtin_amdgcn_exp2f(-2.0f * LOG2E * x)) - 1.0f;
}

#define WMMA_BF16(A, Bm, C) \
  __builtin_amdgcn_wmma_f32_16x16x32_bf16(false, (A), false, (Bm), (short)0, (C), false, false)
// First WMMA of each accumulation chain: C = 0 (folds to inline-0 SRC2).
#define WMMA_BF16_Z(A, Bm) \
  __builtin_amdgcn_wmma_f32_16x16x32_bf16(false, (A), false, (Bm), (short)0, (v8f)0.0f, false, false)

#define BOP(base, g, kk, GS) (*(const v16bf*)((base) + (g) * (GS) + (kk) * KK_STRIDE))

// ---------------------------------------------------------------------------
// Persistent recurrent kernel: grid = 16 (batch tiles of 16), block = 512
// (16 waves). Wave w owns gate-column tile w of each gate block (i,f,g,o);
// c lives in 8 VGPRs per lane for the whole sequence; h is exchanged via LDS.
// Weights stream from L2 every step (768 KB working set, L2 = 192 MB);
// x_t staging is software-pipelined through a double-buffered LDS tile.
// ---------------------------------------------------------------------------
__global__ void __launch_bounds__(512, 1)
lstm_persistent(const float* __restrict__ x,
                const unsigned short* __restrict__ wih_pk,
                const unsigned short* __restrict__ whh_pk,
                const float* __restrict__ b_ih,
                const float* __restrict__ b_hh,
                const unsigned short* __restrict__ wout_pk,
                const float* __restrict__ b_out,
                float* __restrict__ out) {
  // Row-padded LDS tiles: stride % 64 banks == 4 dwords -> conflict-free
  // ds_load_b128 across the 16 M-lanes.
  __shared__ bf16 xs[2][16 * XS_ROW];  // double-buffered x_t tile [16][128+8]
  __shared__ bf16 hs[16 * HS_ROW];     // h tile [16][256+8]

  const int tid  = threadIdx.x;
  const int lane = tid & 31;
  const int wave = tid >> 5;     // 0..15
  const int m    = lane & 15;    // WMMA M row / N col within tile
  const int s    = lane >> 4;    // K-half selector
  const int b0   = blockIdx.x * 16;

  // h0 = 0
  for (int i = tid; i < 16 * HS_ROW; i += 512) hs[i] = (bf16)0.0f;

  // Per-lane fused bias (b_ih + b_hh) for this wave's 4 gate columns;
  // applied at activation time (keeps accumulator init at inline 0).
  float biasv[4];
#pragma unroll
  for (int g = 0; g < 4; ++g) {
    const int col = g * 256 + wave * 16 + m;
    biasv[g] = b_ih[col] + b_hh[col];
  }

  // Per-wave streamed weight base pointers (element units). Gate and k-step
  // selection are immediate offsets off these pointers.
  const unsigned short* wih_w = wih_pk + ((size_t)wave * 4 * 32 + lane) * 16;
  const unsigned short* whh_w = whh_pk + ((size_t)wave * 8 * 32 + lane) * 16;

  // x staging: thread -> (row, 4 consecutive cols)
  const int xrow = tid >> 5;            // 0..15
  const int xcol = (tid & 31) * 4;      // 0..124
  const float* xsrc = x + (size_t)(b0 + xrow) * S_ * I_ + xcol;

  // Prologue: stage x_0 into buffer 0.
  {
    const float4 v = *(const float4*)(xsrc);
    bf16* dst = xs[0] + xrow * XS_ROW + xcol;
    dst[0] = (bf16)v.x; dst[1] = (bf16)v.y;
    dst[2] = (bf16)v.z; dst[3] = (bf16)v.w;
  }

  v8f c = (v8f)0.0f;  // cell state, register-resident for all 512 steps

  const bf16* arow_h = hs + m * HS_ROW;

  // Always-zero laundered offset: blocks LICM of the weight loads (which
  // previously spilled 384 VGPRs of hoisted operands) while keeping the
  // base pointers traceable to kernargs -> global_load_b128, not flat.
  unsigned int woff = 0;

  for (int t = 0; t < S_; ++t) {
    __syncthreads();  // xs[t&1] staged, hs from step t-1 visible

    asm volatile("" : "+v"(woff));
    const unsigned short* wih_t = wih_w + woff;
    const unsigned short* whh_t = whh_w + woff;

    // Pipelined x load for step t+1 (index wraps at S_, always valid);
    // consumed only after the WMMA stream below. Prefetch t+2 into L2.
    const int tn = (t + 1) & (S_ - 1);
    const float4 vx = *(const float4*)(xsrc + (size_t)tn * I_);
    __builtin_prefetch(xsrc + (size_t)((t + 2) & (S_ - 1)) * I_, 0, 1);

    const bf16* arow_x = xs[t & 1] + m * XS_ROW;

    v8f a0, a1, a2, a3;
    // gates = x_t @ W_ih^T : kk = 0 starts each chain with C = 0.
    {
      const v16bf a = load_a_tile(arow_x, 0, s);
      a0 = WMMA_BF16_Z(a, BOP(wih_t, 0, 0, GIH_STRIDE));
      a1 = WMMA_BF16_Z(a, BOP(wih_t, 1, 0, GIH_STRIDE));
      a2 = WMMA_BF16_Z(a, BOP(wih_t, 2, 0, GIH_STRIDE));
      a3 = WMMA_BF16_Z(a, BOP(wih_t, 3, 0, GIH_STRIDE));
    }
#pragma unroll
    for (int kk = 1; kk < 4; ++kk) {
      const v16bf a = load_a_tile(arow_x, kk, s);
      a0 = WMMA_BF16(a, BOP(wih_t, 0, kk, GIH_STRIDE), a0);
      a1 = WMMA_BF16(a, BOP(wih_t, 1, kk, GIH_STRIDE), a1);
      a2 = WMMA_BF16(a, BOP(wih_t, 2, kk, GIH_STRIDE), a2);
      a3 = WMMA_BF16(a, BOP(wih_t, 3, kk, GIH_STRIDE), a3);
    }
    // gates += h @ W_hh^T (K = 256 -> 8 k-steps x 4 gates)
#pragma unroll
    for (int kk = 0; kk < 8; ++kk) {
      const v16bf a = load_a_tile(arow_h, kk, s);
      a0 = WMMA_BF16(a, BOP(whh_t, 0, kk, GHH_STRIDE), a0);
      a1 = WMMA_BF16(a, BOP(whh_t, 1, kk, GHH_STRIDE), a1);
      a2 = WMMA_BF16(a, BOP(whh_t, 2, kk, GHH_STRIDE), a2);
      a3 = WMMA_BF16(a, BOP(whh_t, 3, kk, GHH_STRIDE), a3);
    }

    // Drain the pipelined x_{t+1} into the other LDS buffer (its previous
    // readers finished before the last barrier; next readers start after
    // the next one).
    {
      bf16* dst = xs[tn & 1] + xrow * XS_ROW + xcol;
      dst[0] = (bf16)vx.x; dst[1] = (bf16)vx.y;
      dst[2] = (bf16)vx.z; dst[3] = (bf16)vx.w;
    }

    // Activations + state update (this wave's 16x16 slice; c stays in VGPRs).
    float hval[8];
#pragma unroll
    for (int r = 0; r < 8; ++r) {
      const float iv = sigf(a0[r] + biasv[0]);
      const float fv = sigf(a1[r] + biasv[1]);
      const float gv = tanh_fast(a2[r] + biasv[2]);
      const float ov = sigf(a3[r] + biasv[3]);
      const float cn = fv * c[r] + iv * gv;
      c[r] = cn;
      hval[r] = ov * tanh_fast(cn);
    }

    __syncthreads();  // all waves done reading hs (old h)
    {
      // D layout: lanes 0-15 VGPR r -> M=r; lanes 16-31 -> M=8+r; N = m.
      bf16* hdst = hs + (8 * s) * HS_ROW + wave * 16 + m;
#pragma unroll
      for (int r = 0; r < 8; ++r) hdst[r * HS_ROW] = (bf16)hval[r];
    }
  }
  __syncthreads();

  // out = relu(h_last) @ W_out^T + b_out : 4 N-tiles, waves 0..3.
  if (wave < 4) {
    const float bo = b_out[wave * 16 + m];
    const unsigned short* wbase = wout_pk + ((size_t)wave * 8 * 32 + lane) * 16;
    v8f acc;
    {
      v16bf a = load_a_tile(arow_h, 0, s);
#pragma unroll
      for (int e = 0; e < 16; ++e) {  // ReLU in bf16 domain
        const float v = (float)a[e];
        a[e] = (bf16)(v > 0.0f ? v : 0.0f);
      }
      acc = WMMA_BF16_Z(a, *(const v16bf*)(wbase));
    }
#pragma unroll
    for (int kk = 1; kk < 8; ++kk) {
      v16bf a = load_a_tile(arow_h, kk, s);
#pragma unroll
      for (int e = 0; e < 16; ++e) {
        const float v = (float)a[e];
        a[e] = (bf16)(v > 0.0f ? v : 0.0f);
      }
      acc = WMMA_BF16(a, *(const v16bf*)(wbase + kk * KK_STRIDE), acc);
    }
#pragma unroll
    for (int r = 0; r < 8; ++r)
      out[(size_t)(b0 + r + 8 * s) * O_ + wave * 16 + m] = acc[r] + bo;
  }
}

// ---------------------------------------------------------------------------
extern "C" void kernel_launch(void* const* d_in, const int* in_sizes, int n_in,
                              void* d_out, int out_size, void* d_ws, size_t ws_size,
                              hipStream_t stream) {
  const float* x     = (const float*)d_in[0];
  const float* W_ih  = (const float*)d_in[1];
  const float* W_hh  = (const float*)d_in[2];
  const float* b_ih  = (const float*)d_in[3];
  const float* b_hh  = (const float*)d_in[4];
  const float* W_out = (const float*)d_in[5];
  const float* b_out = (const float*)d_in[6];
  float* out = (float*)d_out;

  char* ws = (char*)d_ws;  // needs ~820 KB
  unsigned short* wih_pk  = (unsigned short*)(ws + WIH_PK_OFF);
  unsigned short* whh_pk  = (unsigned short*)(ws + WHH_PK_OFF);
  unsigned short* wout_pk = (unsigned short*)(ws + WOUT_PK_OFF);

  // Pack weights into bf16 WMMA B-operand order (L2-resident afterwards).
  {
    const int t_ih  = 64 * 4 * 32;   // ntiles * ksteps * 32
    const int t_hh  = 64 * 8 * 32;
    const int t_out = 4 * 8 * 32;
    pack_b_operand<<<(t_ih + 255) / 256, 256, 0, stream>>>(W_ih, wih_pk, 128, 4, t_ih);
    pack_b_operand<<<(t_hh + 255) / 256, 256, 0, stream>>>(W_hh, whh_pk, 256, 8, t_hh);
    pack_b_operand<<<(t_out + 255) / 256, 256, 0, stream>>>(W_out, wout_pk, 256, 8, t_out);
  }

  // 16 batch tiles x 16 waves; entire recurrence in one kernel.
  lstm_persistent<<<dim3(B_ / 16), dim3(512), 0, stream>>>(
      x, wih_pk, whh_pk, b_ih, b_hh, wout_pk, b_out, out);
}
